// SelfComposingPolicyModule_39960375722668
// MI455X (gfx1250) — compile-verified, matrix operand order checked by reading, not run on hardware
//
#include <hip/hip_runtime.h>
#include <hip/hip_bf16.h>
#include <math.h>

typedef __attribute__((ext_vector_type(16))) __bf16 v16bf;
typedef __attribute__((ext_vector_type(8)))  __bf16 v8bf;
typedef __attribute__((ext_vector_type(4)))  __bf16 v4bf;
typedef __attribute__((ext_vector_type(8)))  float  v8f;

#define B_SZ    512
#define D_ENC   1024
#define D_MODEL 1024
#define N_ACT   512
#define N_PREV  256

// ---------------------------------------------------------------------------
// Sinusoidal positional encoding: pe[pos, 2i] = sin(pos*div_i),
// pe[pos, 2i+1] = cos(pos*div_i). One 257-row table serves both heads.
// ---------------------------------------------------------------------------
__global__ __launch_bounds__(256) void pe_fill_kernel(float* __restrict__ pe) {
    int pos = blockIdx.x;          // 0..256
    int i   = threadIdx.x;         // 0..255 -> column pair (2i, 2i+1)
    float div = __expf((float)(2 * i) * (-9.210340371976184f / 512.0f));
    float a = (float)pos * div;
    float2 sc = { __sinf(a), __cosf(a) };
    *(float2*)&pe[(size_t)pos * 512 + 2 * i] = sc;
}

// ---------------------------------------------------------------------------
// WMMA GEMM: C = A(MxK) * B + optional bias / relu.
// BT=0: B is (K,N) row-major.  BT=1: B is (N,K) row-major (C = A*B^T).
// EPI: 0 = none, 1 = +bias, 2 = relu(+bias).
// Block tile 128x64, 256 threads = 8 waves (4 along M x 2 along N),
// each wave computes 32x32 via 2x2 v_wmma_f32_16x16x32_bf16 fragments.
// Software-pipelined: next K-tile's global loads are issued before this
// tile's WMMAs so the load latency hides under ds_load + v_wmma work.
// M % 128 == 0, K % 32 == 0; N guarded (BT path supports ragged N, e.g. 257).
// ---------------------------------------------------------------------------
template <int BT, int EPI>
__global__ __launch_bounds__(256) void gemm_wmma_kernel(
    const float* __restrict__ A, const float* __restrict__ B,
    const float* __restrict__ bias, float* __restrict__ C,
    int M, int N, int K)
{
    __shared__ __align__(16) __bf16 ldsA[128 * 32];   // [m][k]
    __shared__ __align__(16) __bf16 ldsB[64 * 32];    // [n][k]

    const int t    = threadIdx.x;
    const int m0   = blockIdx.y * 128;
    const int n0   = blockIdx.x * 64;
    const int wave = t >> 5;
    const int lane = t & 31;
    const int wm   = (wave & 3) * 32;   // wave row offset in block tile
    const int wn   = (wave >> 2) * 32;  // wave col offset in block tile
    const int lr   = lane & 15;
    const int kh   = lane >> 4;         // K-half selector per ISA layout

    // Per-thread staging coordinates (fixed across K tiles).
    const int ar[4] = { (t + 0) >> 3, (t + 256) >> 3, (t + 512) >> 3, (t + 768) >> 3 };
    const int ac    = (t & 7) * 4;
    // BT=1: B row n / quad c4.  BT=0: B k-row / n quad.
    const int bn0 = (t + 0) >> 3, bn1 = (t + 256) >> 3;       // BT=1
    const int bk0 = (t + 0) >> 4, bk1 = (t + 256) >> 4;       // BT=0
    const int bq  = (t & 15) * 4;                             // BT=0 n offset

    float4 ra[4], rb[2];

    // Issue global loads for K-tile at kk into ra/rb.
    auto load_tiles = [&](int kk) {
        #pragma unroll
        for (int u = 0; u < 4; ++u)
            ra[u] = *(const float4*)&A[(size_t)(m0 + ar[u]) * K + kk + ac];
        if (BT) {
            const int bn[2] = { bn0, bn1 };
            #pragma unroll
            for (int u = 0; u < 2; ++u) {
                float4 x = { 0.f, 0.f, 0.f, 0.f };
                if (n0 + bn[u] < N)
                    x = *(const float4*)&B[(size_t)(n0 + bn[u]) * K + kk + ac];
                rb[u] = x;
            }
        } else {
            const int bk[2] = { bk0, bk1 };
            #pragma unroll
            for (int u = 0; u < 2; ++u) {
                float4 x = { 0.f, 0.f, 0.f, 0.f };
                if (n0 + bq < N)            // N % 4 == 0 on all BT=0 calls
                    x = *(const float4*)&B[(size_t)(kk + bk[u]) * N + n0 + bq];
                rb[u] = x;
            }
        }
    };
    // Convert staged registers to bf16 and write LDS.
    auto store_tiles = [&]() {
        #pragma unroll
        for (int u = 0; u < 4; ++u) {
            v4bf y = { (__bf16)ra[u].x, (__bf16)ra[u].y, (__bf16)ra[u].z, (__bf16)ra[u].w };
            *(v4bf*)&ldsA[ar[u] * 32 + ac] = y;
        }
        if (BT) {
            const int bn[2] = { bn0, bn1 };
            #pragma unroll
            for (int u = 0; u < 2; ++u) {
                v4bf y = { (__bf16)rb[u].x, (__bf16)rb[u].y, (__bf16)rb[u].z, (__bf16)rb[u].w };
                *(v4bf*)&ldsB[bn[u] * 32 + ac] = y;
            }
        } else {
            const int bk[2] = { bk0, bk1 };
            #pragma unroll
            for (int u = 0; u < 2; ++u) {
                ldsB[(bq + 0) * 32 + bk[u]] = (__bf16)rb[u].x;
                ldsB[(bq + 1) * 32 + bk[u]] = (__bf16)rb[u].y;
                ldsB[(bq + 2) * 32 + bk[u]] = (__bf16)rb[u].z;
                ldsB[(bq + 3) * 32 + bk[u]] = (__bf16)rb[u].w;
            }
        }
    };

    v8f acc[2][2] = {};

    load_tiles(0);                       // prologue: tile 0 in flight
    for (int kk = 0; kk < K; kk += 32) {
        store_tiles();                   // waits on outstanding loads here
        __syncthreads();
        if (kk + 32 < K)
            load_tiles(kk + 32);         // next tile's loads fly under WMMAs

        // Fragment assembly per 16-bit A/B VGPR layout:
        // element e <-> K = 16*(e>>3) + 8*(lane>>4) + (e&7); two 16B chunks.
        union Frag { v16bf v; v8bf h[2]; };
        Frag fa[2], fb[2];
        #pragma unroll
        for (int i = 0; i < 2; ++i) {
            const __bf16* rp = &ldsA[(wm + 16 * i + lr) * 32];
            fa[i].h[0] = *(const v8bf*)(rp + kh * 8);
            fa[i].h[1] = *(const v8bf*)(rp + 16 + kh * 8);
            const __bf16* rq = &ldsB[(wn + 16 * i + lr) * 32];
            fb[i].h[0] = *(const v8bf*)(rq + kh * 8);
            fb[i].h[1] = *(const v8bf*)(rq + 16 + kh * 8);
        }
        #pragma unroll
        for (int i = 0; i < 2; ++i)
            #pragma unroll
            for (int j = 0; j < 2; ++j)
                acc[i][j] = __builtin_amdgcn_wmma_f32_16x16x32_bf16(
                    false, fa[i].v, false, fb[j].v, (short)0, acc[i][j], false, false);
        __syncthreads();
    }

    // Epilogue: D layout — VGPR r holds M = r + 8*(lane>>4), N = lane&15.
    #pragma unroll
    for (int i = 0; i < 2; ++i) {
        int row = m0 + wm + 16 * i + 8 * kh;
        #pragma unroll
        for (int j = 0; j < 2; ++j) {
            int col = n0 + wn + 16 * j + lr;
            if (col < N) {
                float bv = (EPI > 0) ? bias[col] : 0.0f;
                #pragma unroll
                for (int r = 0; r < 8; ++r) {
                    float val = acc[i][j][r] + bv;
                    if (EPI == 2) val = fmaxf(val, 0.0f);
                    C[(size_t)(row + r) * N + col] = val;
                }
            }
        }
    }
}

// ---------------------------------------------------------------------------
// Phi pass 1 (HBM-bound): per (b,n) compute qk_out . phi[b,n] and
// qk_in . phi[b,n], add pe-dot terms, scale. Block per b, wave per n.
// qk vectors held in registers (8 x float4) so the loop is loads + FMAs.
// ---------------------------------------------------------------------------
__global__ __launch_bounds__(256) void scores_phi_kernel(
    const float* __restrict__ phi,
    const float* __restrict__ qk_out, const float* __restrict__ qk_in,
    const float* __restrict__ pe_dot_out, const float* __restrict__ pe_dot_in,
    float* __restrict__ scores_out, float* __restrict__ scores_in,
    float inv_sqrt)
{
    int b = blockIdx.x;
    int wave = threadIdx.x >> 5, lane = threadIdx.x & 31;
    const float4* qo4 = (const float4*)(qk_out + (size_t)b * N_ACT);
    const float4* qi4 = (const float4*)(qk_in  + (size_t)b * N_ACT);
    float4 a[4], c[4];
    #pragma unroll
    for (int j = 0; j < 4; ++j) {
        a[j] = qo4[lane + 32 * j];
        c[j] = qi4[lane + 32 * j];
    }
    for (int n = wave * 32; n < wave * 32 + 32; ++n) {
        const float4* p4 = (const float4*)(phi + ((size_t)b * N_PREV + n) * N_ACT);
        float4 x[4];
        #pragma unroll
        for (int j = 0; j < 4; ++j) x[j] = p4[lane + 32 * j];   // batched b128
        float s1 = 0.0f, s2 = 0.0f;
        #pragma unroll
        for (int j = 0; j < 4; ++j) {
            s1 += x[j].x * a[j].x + x[j].y * a[j].y + x[j].z * a[j].z + x[j].w * a[j].w;
            s2 += x[j].x * c[j].x + x[j].y * c[j].y + x[j].z * c[j].z + x[j].w * c[j].w;
        }
        #pragma unroll
        for (int off = 16; off; off >>= 1) {
            s1 += __shfl_xor(s1, off, 32);
            s2 += __shfl_xor(s2, off, 32);
        }
        if (lane == 0) {
            scores_out[(size_t)b * N_PREV + n] =
                (s1 + pe_dot_out[(size_t)b * N_PREV + n]) * inv_sqrt;
            scores_in[(size_t)b * 257 + 1 + n] =
                (s2 + pe_dot_in[(size_t)b * 257 + 1 + n]) * inv_sqrt;
        }
    }
}

// ---------------------------------------------------------------------------
// Generic row softmax, optional elementwise add (final softmax(v+logits)).
// One block (256 threads) per row; cols <= 512.
// ---------------------------------------------------------------------------
__global__ __launch_bounds__(256) void softmax_kernel(
    const float* __restrict__ src, const float* __restrict__ add,
    float* __restrict__ dst, int cols, int ld)
{
    int b = blockIdx.x, t = threadIdx.x;
    __shared__ float red[256];
    int c0 = t, c1 = t + 256;
    float x0 = (c0 < cols) ? src[(size_t)b * ld + c0] + (add ? add[(size_t)b * ld + c0] : 0.0f)
                           : -INFINITY;
    float x1 = (c1 < cols) ? src[(size_t)b * ld + c1] + (add ? add[(size_t)b * ld + c1] : 0.0f)
                           : -INFINITY;
    red[t] = fmaxf(x0, x1);
    __syncthreads();
    for (int s = 128; s; s >>= 1) { if (t < s) red[t] = fmaxf(red[t], red[t + s]); __syncthreads(); }
    float mx = red[0];
    __syncthreads();
    float e0 = (c0 < cols) ? __expf(x0 - mx) : 0.0f;
    float e1 = (c1 < cols) ? __expf(x1 - mx) : 0.0f;
    red[t] = e0 + e1;
    __syncthreads();
    for (int s = 128; s; s >>= 1) { if (t < s) red[t] += red[t + s]; __syncthreads(); }
    float inv = 1.0f / red[0];
    if (c0 < cols) dst[(size_t)b * ld + c0] = e0 * inv;
    if (c1 < cols) dst[(size_t)b * ld + c1] = e1 * inv;
}

// ---------------------------------------------------------------------------
// Phi pass 2: v[b] = attn_out[b] @ phi[b]; scores_in[b,0] =
// (qk_in[b].v[b] + pe_dot_in[b,0]) * inv_sqrt. 128 threads, float4/thread.
// ---------------------------------------------------------------------------
__global__ __launch_bounds__(128) void v_kernel(
    const float* __restrict__ phi, const float* __restrict__ attn_out,
    const float* __restrict__ qk_in, const float* __restrict__ pe_dot_in,
    float* __restrict__ vout, float* __restrict__ scores_in, float inv_sqrt)
{
    int b = blockIdx.x, t = threadIdx.x;   // t < 128
    __shared__ float aw[N_PREV];
    __shared__ float red[128];
    aw[t]       = attn_out[(size_t)b * N_PREV + t];
    aw[t + 128] = attn_out[(size_t)b * N_PREV + t + 128];
    __syncthreads();
    const float4* pb4 = (const float4*)(phi + (size_t)b * N_PREV * N_ACT);
    float4 acc = { 0.f, 0.f, 0.f, 0.f };
    for (int n = 0; n < N_PREV; ++n) {     // per n: 128 lanes x 16B = 2KB row
        if (n + 4 < N_PREV)
            __builtin_prefetch(&pb4[(size_t)(n + 4) * 128 + t], 0, 0);
        float w = aw[n];
        float4 x = pb4[(size_t)n * 128 + t];
        acc.x += w * x.x; acc.y += w * x.y; acc.z += w * x.z; acc.w += w * x.w;
    }
    ((float4*)vout)[(size_t)b * 128 + t] = acc;
    float4 q = ((const float4*)qk_in)[(size_t)b * 128 + t];
    red[t] = acc.x * q.x + acc.y * q.y + acc.z * q.z + acc.w * q.w;
    __syncthreads();
    for (int s = 64; s; s >>= 1) { if (t < s) red[t] += red[t + s]; __syncthreads(); }
    if (t == 0)
        scores_in[(size_t)b * 257] = (red[0] + pe_dot_in[(size_t)b * 257]) * inv_sqrt;
}

// ---------------------------------------------------------------------------
// Phi pass 3: pbar[b] = attn_in[b,0]*v[b] + sum_n attn_in[b,n+1]*phi[b,n].
// ---------------------------------------------------------------------------
__global__ __launch_bounds__(128) void pbar_kernel(
    const float* __restrict__ phi, const float* __restrict__ attn_in,
    const float* __restrict__ vin, float* __restrict__ pbar)
{
    int b = blockIdx.x, t = threadIdx.x;   // t < 128
    __shared__ float aw[257];
    aw[t]       = attn_in[(size_t)b * 257 + t];
    aw[t + 128] = attn_in[(size_t)b * 257 + t + 128];
    if (t == 0) aw[256] = attn_in[(size_t)b * 257 + 256];
    __syncthreads();
    float a0 = aw[0];
    float4 v4 = ((const float4*)vin)[(size_t)b * 128 + t];
    float4 acc = { a0 * v4.x, a0 * v4.y, a0 * v4.z, a0 * v4.w };
    const float4* pb4 = (const float4*)(phi + (size_t)b * N_PREV * N_ACT);
    for (int n = 0; n < N_PREV; ++n) {
        if (n + 4 < N_PREV)
            __builtin_prefetch(&pb4[(size_t)(n + 4) * 128 + t], 0, 0);
        float w = aw[n + 1];
        float4 x = pb4[(size_t)n * 128 + t];
        acc.x += w * x.x; acc.y += w * x.y; acc.z += w * x.z; acc.w += w * x.w;
    }
    ((float4*)pbar)[(size_t)b * 128 + t] = acc;
}

// x = [h_s ; ctx] per row (float4 copies).
__global__ __launch_bounds__(256) void concat_kernel(
    const float* __restrict__ h_s, const float* __restrict__ ctx, float* __restrict__ x)
{
    int b = blockIdx.x, t = threadIdx.x;
    float4* x4 = (float4*)x;
    x4[(size_t)b * 512 + t]       = ((const float4*)h_s)[(size_t)b * 256 + t];
    x4[(size_t)b * 512 + 256 + t] = ((const float4*)ctx)[(size_t)b * 256 + t];
}

// ---------------------------------------------------------------------------
extern "C" void kernel_launch(void* const* d_in, const int* in_sizes, int n_in,
                              void* d_out, int out_size, void* d_ws, size_t ws_size,
                              hipStream_t stream) {
    const float* h_s    = (const float*)d_in[0];
    const float* phi    = (const float*)d_in[1];
    const float* WQ_out = (const float*)d_in[2];
    const float* WK_out = (const float*)d_in[3];
    const float* WQ_in  = (const float*)d_in[4];
    const float* WK_in  = (const float*)d_in[5];
    const float* WV_in  = (const float*)d_in[6];
    const float* W1     = (const float*)d_in[7];
    const float* b1     = (const float*)d_in[8];
    const float* W2     = (const float*)d_in[9];
    const float* b2     = (const float*)d_in[10];
    float* out = (float*)d_out;

    float* ws      = (float*)d_ws;
    float* pe      = ws;                  // 257*512
    float* q_tmp   = pe      + 257 * 512; // 512*1024 (reused for q_out then q_in)
    float* qk_out  = q_tmp   + 512 * 1024;// 512*512
    float* qk_in   = qk_out  + 512 * 512; // 512*512
    float* pe_do   = qk_in   + 512 * 512; // 512*256
    float* pe_di   = pe_do   + 512 * 256; // 512*257
    float* sc_out  = pe_di   + 512 * 257; // 512*256
    float* at_out  = sc_out  + 512 * 256; // 512*256
    float* sc_in   = at_out  + 512 * 256; // 512*257
    float* at_in   = sc_in   + 512 * 257; // 512*257
    float* vbuf    = at_in   + 512 * 257; // 512*512
    float* pbar    = vbuf    + 512 * 512; // 512*512
    float* ctx     = pbar    + 512 * 512; // 512*1024
    float* xcat    = ctx     + 512 * 1024;// 512*2048
    float* hbuf    = xcat    + 512 * 2048;// 512*1024
    float* logits  = hbuf    + 512 * 1024;// 512*512

    const float inv_sqrt = 0.03125f; // 1/sqrt(1024)
    const dim3 blk(256);

    pe_fill_kernel<<<257, blk, 0, stream>>>(pe);

    // qk_out = (h_s @ WQ_out) @ WK_out^T
    gemm_wmma_kernel<0, 0><<<dim3(16, 4), blk, 0, stream>>>(h_s, WQ_out, nullptr, q_tmp, 512, 1024, 1024);
    gemm_wmma_kernel<1, 0><<<dim3(8, 4),  blk, 0, stream>>>(q_tmp, WK_out, nullptr, qk_out, 512, 512, 1024);
    // qk_in = (h_s @ WQ_in) @ WK_in^T
    gemm_wmma_kernel<0, 0><<<dim3(16, 4), blk, 0, stream>>>(h_s, WQ_in, nullptr, q_tmp, 512, 1024, 1024);
    gemm_wmma_kernel<1, 0><<<dim3(8, 4),  blk, 0, stream>>>(q_tmp, WK_in, nullptr, qk_in, 512, 512, 1024);
    // pe-dot terms: qk @ pe^T
    gemm_wmma_kernel<1, 0><<<dim3(4, 4),  blk, 0, stream>>>(qk_out, pe, nullptr, pe_do, 512, 256, 512);
    gemm_wmma_kernel<1, 0><<<dim3(5, 4),  blk, 0, stream>>>(qk_in,  pe, nullptr, pe_di, 512, 257, 512);

    // phi pass 1: both heads' scores
    scores_phi_kernel<<<512, blk, 0, stream>>>(phi, qk_out, qk_in, pe_do, pe_di, sc_out, sc_in, inv_sqrt);
    softmax_kernel<<<512, blk, 0, stream>>>(sc_out, nullptr, at_out, 256, 256);
    // phi pass 2: v + scores_in[:,0]
    v_kernel<<<512, dim3(128), 0, stream>>>(phi, at_out, qk_in, pe_di, vbuf, sc_in, inv_sqrt);
    softmax_kernel<<<512, blk, 0, stream>>>(sc_in, nullptr, at_in, 257, 257);
    // phi pass 3: pbar
    pbar_kernel<<<512, dim3(128), 0, stream>>>(phi, at_in, vbuf, pbar);

    // ctx = pbar @ WV_in
    gemm_wmma_kernel<0, 0><<<dim3(16, 4), blk, 0, stream>>>(pbar, WV_in, nullptr, ctx, 512, 1024, 512);
    // MLP
    concat_kernel<<<512, blk, 0, stream>>>(h_s, ctx, xcat);
    gemm_wmma_kernel<0, 2><<<dim3(16, 4), blk, 0, stream>>>(xcat, W1, b1, hbuf, 512, 1024, 2048);
    gemm_wmma_kernel<0, 1><<<dim3(8, 4),  blk, 0, stream>>>(hbuf, W2, b2, logits, 512, 512, 1024);
    // out = softmax(v + logits)
    softmax_kernel<<<512, blk, 0, stream>>>(vbuf, logits, out, 512, 512);
}
